// YOLOv11Postprocessor_26542897889478
// MI455X (gfx1250) — matrix operand order repaired and checked by reading, not compiled, not run on hardware
//
#include <hip/hip_runtime.h>
#include <hip/hip_bf16.h>

// ---------------------------------------------------------------------------
// YOLOv11 post-process for MI455X (gfx1250, wave32).
//   decode (bandwidth bound, WMMA for DFL integral) -> bitonic top-1024 ->
//   bitmask greedy NMS (single wave32 scan) -> compacted outputs.
// ---------------------------------------------------------------------------

typedef __attribute__((ext_vector_type(16))) _Float16 v16h;
typedef __attribute__((ext_vector_type(8)))  float    v8f;

#define RBINS 16
#define NCLS  80
#define CHN   144       // 4*R + C
#define NBATCH 16
#define ATOT  8400      // 6400 + 1600 + 400
#define KPRE  1024
#define MAXD  300
#define CONF_T 0.25f
#define IOU_T  0.45f
#define CLS_OFF 4096.0f

// ---------------------------------------------------------------------------
// Kernel 1: per-level decode. One wave32 per 16 anchors.
//   DFL expectation via v_wmma_f32_16x16x32_f16:
//     A (16 rows = (anchor,side), K=16 bins padded to 32)  = exp(x - max)
//     B (32x16) = bin index k for K<16, 0 for K>=16  (all columns equal)
//     C col0    = softmax numerator; divide by row-sum afterwards.
//   A-frag wave32 layout: lane<16 holds row m=lane, K=0..7 in halves 0..7
//   (halves 8..15 are K=16..23 -> zero); lane>=16 holds row m=lane-16,
//   K=8..15 in halves 0..7.  Row (max,sum) reductions therefore pair lanes
//   L and L^16 -> __shfl_xor(...,16).
// ---------------------------------------------------------------------------
__global__ __launch_bounds__(32)
void yolo_decode_kernel(const float* __restrict__ p,
                        int H, int W, int stride, int lvl_off,
                        const int* __restrict__ hptr, const int* __restrict__ wptr,
                        float* __restrict__ ws_boxes,
                        float* __restrict__ ws_smax,
                        int*   __restrict__ ws_label)
{
    __shared__ float lds_box[64][16];  // [box channel][anchor]
    __shared__ float lds_num[64];      // DFL numerators, row = a_local*4+side
    __shared__ float lds_den[64];      // softmax denominators

    const int HW   = H * W;
    const int b    = blockIdx.y;
    const int a0   = blockIdx.x * 16;
    const int lane = threadIdx.x;      // 0..31 (wave32)
    const float* pb = p + (size_t)b * CHN * HW;

    // ---- stage 64 DFL channels x 16 anchors into LDS (coalesced 64B rows) --
    {
        const int c2 = lane >> 4;          // 0..1 : two channels per pass
        const int al = lane & 15;
        for (int ch = c2; ch < 64; ch += 2)
            lds_box[ch][al] = pb[(size_t)ch * HW + a0 + al];
    }
    __syncthreads();

    // prefetch class-score region for this tile (global_prefetch_b8)
    __builtin_prefetch(pb + (size_t)64 * HW + a0 + (lane & 15), 0, 1);

    // ---- B fragment: bins 0..15 on K<16 (halves 0..7, K=8*(lane/16)+j), zero on K>=16 --
    v16h bf;
    for (int j = 0; j < 16; ++j)
        bf[j] = (j < 8) ? (_Float16)(float)(8 * (lane >> 4) + j) : (_Float16)0.0f;

    // ---- 4 WMMAs cover the 64 (anchor,side) rows of this tile --------------
    for (int w = 0; w < 4; ++w) {
        const int m    = lane & 15;        // row within this WMMA
        const int row  = w * 16 + m;       // tile row = a_local*4 + side
        const int al   = row >> 2;
        const int side = row & 3;
        const int k0   = (lane < 16) ? 0 : 8;

        float e[8];
        float mx = -1e30f;
        for (int j = 0; j < 8; ++j) {
            float x = lds_box[side * 16 + k0 + j][al];
            e[j] = x;
            mx = fmaxf(mx, x);
        }
        mx = fmaxf(mx, __shfl_xor(mx, 16, 32));      // pair-lane row max
        float sum = 0.0f;
        for (int j = 0; j < 8; ++j) {
            e[j] = __builtin_expf(e[j] - mx);
            sum += e[j];
        }
        sum += __shfl_xor(sum, 16, 32);              // pair-lane row sum

        v16h af;
        for (int j = 0; j < 8;  ++j) af[j] = (_Float16)e[j];
        for (int j = 8; j < 16; ++j) af[j] = (_Float16)0.0f;

        v8f acc = {};
        acc = __builtin_amdgcn_wmma_f32_16x16x32_f16(
                  false, af, false, bf, (short)0, acc, false, false);

        // C layout: VGPR r -> (M=r, N=lane) lanes 0-15 ; (M=8+r) lanes 16-31.
        // All columns are identical, so columns 0 (lane 0 / lane 16) suffice.
        if (lane == 0)
            for (int r = 0; r < 8; ++r) lds_num[w * 16 + r] = acc[r];
        if (lane == 16)
            for (int r = 0; r < 8; ++r) lds_num[w * 16 + 8 + r] = acc[r];
        if (lane < 16)
            lds_den[w * 16 + lane] = sum;
    }
    __syncthreads();

    // ---- assemble boxes: one anchor per lane (lanes 0..15) -----------------
    const int hImg = *hptr, wImg = *wptr;
    if (lane < 16) {
        const int al = lane;
        const int aG = a0 + al;            // anchor index within level
        float d[4];
        for (int s = 0; s < 4; ++s)
            d[s] = lds_num[al * 4 + s] / lds_den[al * 4 + s] * (float)stride;
        float cx = ((float)(aG % W) + 0.5f) * (float)stride;
        float cy = ((float)(aG / W) + 0.5f) * (float)stride;
        float xm = (float)(wImg - 1), ym = (float)(hImg - 1);
        float x1 = fminf(fmaxf(cx - d[0], 0.0f), xm);
        float y1 = fminf(fmaxf(cy - d[1], 0.0f), ym);
        float x2 = fminf(fmaxf(cx + d[2], 0.0f), xm);
        float y2 = fminf(fmaxf(cy + d[3], 0.0f), ym);
        size_t o = ((size_t)b * ATOT + lvl_off + aG) * 4;
        ws_boxes[o + 0] = x1; ws_boxes[o + 1] = y1;
        ws_boxes[o + 2] = x2; ws_boxes[o + 3] = y2;
    }

    // ---- class scores: 80 channels split 40/40 across the lane pair --------
    {
        const int al = lane & 15;
        const int hf = lane >> 4;
        float best = -1e30f;
        int   bl   = hf * 40;
        const float* ps = pb + (size_t)(64 + hf * 40) * HW + a0 + al;
        for (int c = 0; c < 40; ++c) {
            float v = ps[(size_t)c * HW];
            if (v > best) { best = v; bl = hf * 40 + c; }
        }
        float ob = __shfl_xor(best, 16, 32);
        int   ol = __shfl_xor(bl,   16, 32);
        // argmax keeps the lowest class index on ties
        if (ob > best || (ob == best && ol < bl)) { best = ob; bl = ol; }
        if (lane < 16) {
            float sm = 1.0f / (1.0f + __builtin_expf(-best));
            size_t o = (size_t)b * ATOT + lvl_off + a0 + al;
            ws_smax[o]  = sm;
            ws_label[o] = bl;
        }
    }
}

// ---------------------------------------------------------------------------
// helpers: monotone float <-> ordered-uint
// ---------------------------------------------------------------------------
__device__ __forceinline__ unsigned f2ord(float f) {
    unsigned u = __float_as_uint(f);
    return (u & 0x80000000u) ? ~u : (u | 0x80000000u);
}

// ---------------------------------------------------------------------------
// Kernel 2: per-batch top-1024 by full bitonic sort of 16384 padded keys in
// 128 KB dynamic LDS (CDNA5: 320 KB per workgroup). Descending order; index
// stored inverted in the low word so ties break toward the lowest index.
// ---------------------------------------------------------------------------
__global__ __launch_bounds__(1024)
void yolo_topk_kernel(const float* __restrict__ ws_boxes,
                      const float* __restrict__ ws_smax,
                      const int*   __restrict__ ws_label,
                      float* __restrict__ selb,
                      float* __restrict__ sels,
                      int*   __restrict__ sell,
                      int*   __restrict__ selv)
{
    extern __shared__ unsigned long long keys[];   // 16384 entries
    const int b = blockIdx.x;
    const int t = threadIdx.x;
    const int N = 16384;

    for (int i = t; i < N; i += 1024) {
        unsigned long long k = 0ull;               // pad -> sorts last
        if (i < ATOT) {
            float s  = ws_smax[(size_t)b * ATOT + i];
            float se = (s > CONF_T) ? s : -1.0f;   // reference conf masking
            k = ((unsigned long long)f2ord(se) << 32)
              | (unsigned long long)(0xFFFFFFFFu - (unsigned)i);
        }
        keys[i] = k;
    }
    __syncthreads();

    for (int k2 = 2; k2 <= N; k2 <<= 1) {
        for (int j = k2 >> 1; j > 0; j >>= 1) {
            for (int ii = t; ii < N; ii += 1024) {
                int pp = ii ^ j;
                if (pp > ii) {
                    unsigned long long a = keys[ii], c = keys[pp];
                    bool desc = ((ii & k2) == 0);
                    bool sw   = desc ? (a < c) : (a > c);
                    if (sw) { keys[ii] = c; keys[pp] = a; }
                }
            }
            __syncthreads();
        }
    }

    // emit top-1024 (blockDim.x == 1024: one per thread)
    unsigned long long key = keys[t];
    unsigned u    = (unsigned)(key >> 32);
    unsigned bits = (u & 0x80000000u) ? (u & 0x7FFFFFFFu) : ~u;
    float score   = __uint_as_float(bits);
    unsigned idx  = 0xFFFFFFFFu - (unsigned)(key & 0xFFFFFFFFu);
    if (idx >= ATOT) idx = 0;                      // pad entries (never valid)

    size_t so = (size_t)b * KPRE + t;
    size_t bo = ((size_t)b * ATOT + idx) * 4;
    selb[so * 4 + 0] = ws_boxes[bo + 0];
    selb[so * 4 + 1] = ws_boxes[bo + 1];
    selb[so * 4 + 2] = ws_boxes[bo + 2];
    selb[so * 4 + 3] = ws_boxes[bo + 3];
    sels[so] = score;
    sell[so] = ws_label[(size_t)b * ATOT + idx];
    selv[so] = (score > CONF_T) ? 1 : 0;
}

// ---------------------------------------------------------------------------
// Kernel 3: per-batch class-aware greedy NMS.
//   Phase A: all 1024 threads build the 1024x1024 IoU>T bitmask in LDS
//            (stride-33 padding -> conflict-free scan reads).
//   Phase B: wave 0 runs the serial scan; lane w owns suppression word w;
//            the candidate's suppression bit is broadcast with __shfl.
//   Phase C: popcount prefix-sum compaction into the 300 output slots.
// ---------------------------------------------------------------------------
__global__ __launch_bounds__(1024)
void yolo_nms_kernel(const float* __restrict__ selb,
                     const float* __restrict__ sels,
                     const int*   __restrict__ sell,
                     const int*   __restrict__ selv,
                     float* __restrict__ out)
{
    extern __shared__ unsigned lds[];
    unsigned* msk = lds;                           // [1024][33] u32
    float*    bx  = (float*)(lds + 1024 * 33);     // [1024][4] offset boxes
    unsigned* vld = lds + 1024 * 33 + 4096;        // [1024]
    unsigned* kw  = lds + 1024 * 33 + 4096 + 1024; // [32] keep words

    const int b = blockIdx.x;
    const int t = threadIdx.x;

    // load class-offset boxes + valid flags
    {
        size_t so = (size_t)b * KPRE + t;
        float off = (float)sell[so] * CLS_OFF;
        bx[t * 4 + 0] = selb[so * 4 + 0] + off;
        bx[t * 4 + 1] = selb[so * 4 + 1] + off;
        bx[t * 4 + 2] = selb[so * 4 + 2] + off;
        bx[t * 4 + 3] = selb[so * 4 + 3] + off;
        vld[t] = (unsigned)selv[so];
    }
    __syncthreads();

    // Phase A: IoU bitmask row t
    {
        float x1 = bx[t * 4 + 0], y1 = bx[t * 4 + 1];
        float x2 = bx[t * 4 + 2], y2 = bx[t * 4 + 3];
        float ar = (x2 - x1) * (y2 - y1);
        for (int w = 0; w < 32; ++w) {
            unsigned bits = 0;
            for (int q = 0; q < 32; ++q) {
                int j = w * 32 + q;
                float jx1 = bx[j * 4 + 0], jy1 = bx[j * 4 + 1];
                float jx2 = bx[j * 4 + 2], jy2 = bx[j * 4 + 3];
                float iw = fmaxf(fminf(x2, jx2) - fmaxf(x1, jx1), 0.0f);
                float ih = fmaxf(fminf(y2, jy2) - fmaxf(y1, jy1), 0.0f);
                float inter = iw * ih;
                float arj = (jx2 - jx1) * (jy2 - jy1);
                float iou = inter / (ar + arj - inter + 1e-7f);
                if (iou > IOU_T && j != t) bits |= (1u << q);
            }
            msk[t * 33 + w] = bits;
        }
    }
    __syncthreads();

    // Phase B: serial greedy scan on one wave32
    if (t < 32) {
        unsigned supp = 0u, keep = 0u;
        for (int i = 0; i < 1024; ++i) {
            int owner = i >> 5;
            unsigned sw = __shfl(supp, owner, 32);
            bool kept = !((sw >> (i & 31)) & 1u) && (vld[i] != 0u);
            if (kept) {
                supp |= msk[i * 33 + t];
                if (t == owner) keep |= (1u << (i & 31));
            }
        }
        kw[t] = keep;
    }
    __syncthreads();

    // output pointer partitions: boxes | scores | labels | valid
    float* ob = out;                               // [16][300][4]
    float* os = out + NBATCH * MAXD * 4;           // [16][300]
    float* ol = out + NBATCH * MAXD * 5;
    float* ov = out + NBATCH * MAXD * 6;

    // Phase C1: defaults + valid flags
    if (t < MAXD) {
        int n = 0;
        for (int w = 0; w < 32; ++w) n += __popc(kw[w]);
        if (n > MAXD) n = MAXD;
        size_t o = (size_t)b * MAXD + t;
        ob[o * 4 + 0] = 0.0f; ob[o * 4 + 1] = 0.0f;
        ob[o * 4 + 2] = 0.0f; ob[o * 4 + 3] = 0.0f;
        os[o] = 0.0f;
        ol[o] = -1.0f;
        ov[o] = (t < n) ? 1.0f : 0.0f;
    }
    __syncthreads();

    // Phase C2: scatter kept detections by rank
    {
        int w = t >> 5, q = t & 31;
        if ((kw[w] >> q) & 1u) {
            int rank = 0;
            for (int x = 0; x < w; ++x) rank += __popc(kw[x]);
            rank += __popc(kw[w] & ((1u << q) - 1u));
            if (rank < MAXD) {
                size_t o  = (size_t)b * MAXD + rank;
                size_t so = (size_t)b * KPRE + t;
                ob[o * 4 + 0] = selb[so * 4 + 0];  // un-offset clipped boxes
                ob[o * 4 + 1] = selb[so * 4 + 1];
                ob[o * 4 + 2] = selb[so * 4 + 2];
                ob[o * 4 + 3] = selb[so * 4 + 3];
                os[o] = sels[so];
                ol[o] = (float)sell[so];
            }
        }
    }
}

// ---------------------------------------------------------------------------
extern "C" void kernel_launch(void* const* d_in, const int* in_sizes, int n_in,
                              void* d_out, int out_size, void* d_ws, size_t ws_size,
                              hipStream_t stream) {
    const float* p0   = (const float*)d_in[0];   // (16,144,80,80)
    const float* p1   = (const float*)d_in[1];   // (16,144,40,40)
    const float* p2   = (const float*)d_in[2];   // (16,144,20,20)
    const int*   hptr = (const int*)d_in[3];
    const int*   wptr = (const int*)d_in[4];

    // workspace layout (floats): ~3.7 MB total
    float* ws       = (float*)d_ws;
    float* ws_boxes = ws;                          // 16*8400*4 = 537600
    float* ws_smax  = ws + 537600;                 // 16*8400   = 134400
    int*   ws_label = (int*)(ws + 672000);         // 16*8400   = 134400
    float* selb     = ws + 806400;                 // 16*1024*4 =  65536
    float* sels     = ws + 871936;                 // 16*1024   =  16384
    int*   sell     = (int*)(ws + 888320);         // 16*1024
    int*   selv     = (int*)(ws + 904704);         // 16*1024

    dim3 blk(32, 1, 1);
    yolo_decode_kernel<<<dim3(6400 / 16, NBATCH), blk, 0, stream>>>(
        p0, 80, 80,  8,    0, hptr, wptr, ws_boxes, ws_smax, ws_label);
    yolo_decode_kernel<<<dim3(1600 / 16, NBATCH), blk, 0, stream>>>(
        p1, 40, 40, 16, 6400, hptr, wptr, ws_boxes, ws_smax, ws_label);
    yolo_decode_kernel<<<dim3( 400 / 16, NBATCH), blk, 0, stream>>>(
        p2, 20, 20, 32, 8000, hptr, wptr, ws_boxes, ws_smax, ws_label);

    yolo_topk_kernel<<<NBATCH, 1024, 16384 * sizeof(unsigned long long), stream>>>(
        ws_boxes, ws_smax, ws_label, selb, sels, sell, selv);

    size_t nms_lds = (size_t)(1024 * 33 + 4096 + 1024 + 32) * sizeof(unsigned);
    yolo_nms_kernel<<<NBATCH, 1024, nms_lds, stream>>>(
        selb, sels, sell, selv, (float*)d_out);
}
